// AdvancedEGFRNet_23218593202315
// MI455X (gfx1250) — matrix operand order repaired and checked by reading, not compiled, not run on hardware
//
#include <hip/hip_runtime.h>
#include <hip/hip_bf16.h>
#include <cstring>

typedef __attribute__((ext_vector_type(16))) _Float16 v16h;
typedef __attribute__((ext_vector_type(8)))  _Float16 v8h;
typedef __attribute__((ext_vector_type(8)))  float    v8f;

#define HID 256
#define HEADS 8
#define CCH 32
#define FEDGE 16
#define LAYERS 4
#define SLOPE 0.2f
#define EPS 1e-5f

// B staged transposed in LDS with padded stride: 16B-aligned rows, 16-bank spread
#define BT_STRIDE 88

// ---------- helpers ----------
__device__ __forceinline__ unsigned f32_enc(float f) {
    unsigned u = __float_as_uint(f);
    return (u & 0x80000000u) ? ~u : (u | 0x80000000u);
}
__device__ __forceinline__ float f32_dec(unsigned u) {
    return __uint_as_float((u & 0x80000000u) ? (u ^ 0x80000000u) : ~u);
}
__device__ __forceinline__ float lrelu(float x) { return x > 0.f ? x : SLOPE * x; }

__global__ void k_zero_f32(float* p, long long n) {
    long long i = (long long)blockIdx.x * blockDim.x + threadIdx.x;
    if (i < n) p[i] = 0.f;
}
__global__ void k_fill_u32(unsigned* p, unsigned v, long long n) {
    long long i = (long long)blockIdx.x * blockDim.x + threadIdx.x;
    if (i < n) p[i] = v;
}
__global__ void k_f32_to_f16(const float* x, _Float16* y, long long n) {
    long long i = (long long)blockIdx.x * blockDim.x + threadIdx.x;
    if (i < n) y[i] = (_Float16)x[i];
}

// ---------- generic WMMA GEMM: Y[M,Ncols] = A[M,K](f16) @ B[K,Ncols](f16) (+bias) ----------
// block = 256 threads = 8 waves; block tile = 32 rows x 64 cols
//   wave w: rows (w>>2)*16..+15, cols (w&3)*16..+15
// K chunked by 64 in LDS; A row-major, B transposed (col-major, padded stride).
// K multiple of 64, Ncols multiple of 64, rows guarded (M arbitrary).
__global__ void k_gemm_wmma_f16(const _Float16* __restrict__ A,
                                const _Float16* __restrict__ B,
                                const float* __restrict__ bias,
                                float* __restrict__ Y,
                                int M, int K, int Ncols) {
    __shared__ _Float16 sA[32 * 64];
    __shared__ _Float16 sBt[64 * BT_STRIDE];
    const int tid   = threadIdx.x;
    const int lane  = tid & 31;
    const int wave  = tid >> 5;
    const int hi    = (lane >= 16) ? 1 : 0;
    const int lrow  = lane & 15;
    const int rtile = (wave >> 2) * 16;              // 0 or 16 within block tile
    const int ctile = (wave & 3) * 16;               // 0/16/32/48 within block tile
    const int m0    = blockIdx.x * 32;
    const int nblk  = blockIdx.y * 64;

    // staging coordinates (vectorized 16B segments)
    const int ar  = tid >> 3;            // A: row 0..31
    const int asg = (tid & 7) * 8;       // A: 8-half segment
    v8f acc = {};
    for (int kc = 0; kc < K; kc += 64) {
        // stage A chunk: one guarded b128 load + b128 LDS store per thread
        {
            int row = m0 + ar;
            v8h av = {};
            if (row < M) av = *(const v8h*)(A + (long long)row * K + kc + asg);
            *(v8h*)(sA + ar * 64 + asg) = av;
        }
        // stage B chunk transposed: b128 global loads, strided b16 LDS stores
        #pragma unroll
        for (int it = 0; it < 2; ++it) {
            int slot = tid + it * 256;
            int r  = slot >> 3;          // k-row 0..63
            int sg = (slot & 7) * 8;     // col segment
            v8h bv = *(const v8h*)(B + (long long)(kc + r) * Ncols + nblk + sg);
            #pragma unroll
            for (int j = 0; j < 8; ++j) sBt[(sg + j) * BT_STRIDE + r] = bv[j];
        }
        __syncthreads();
        #pragma unroll
        for (int k0 = 0; k0 < 64; k0 += 32) {
            union { v16h v; _Float16 h[16]; } a, b;
            // A 16x32 f16 fragment (ISA 7.12.2): two contiguous 8-half runs -> ds_load_b128
            const int abase = (rtile + lrow) * 64 + k0 + (hi ? 8 : 0);
            #pragma unroll
            for (int j = 0; j < 8; ++j) a.h[j]     = sA[abase + j];
            #pragma unroll
            for (int j = 0; j < 8; ++j) a.h[8 + j] = sA[abase + 16 + j];
            // B 32x16 fragment: for fixed column, two 8-half K runs (same order as A) -> 2x ds_load_b128
            const int bbase = (ctile + lrow) * BT_STRIDE + k0 + (hi ? 8 : 0);
            #pragma unroll
            for (int j = 0; j < 16; ++j) b.h[j] = sBt[bbase + j + (j >= 8 ? 8 : 0)];
            acc = __builtin_amdgcn_wmma_f32_16x16x32_f16(false, a.v, false, b.v,
                                                         (short)0, acc, false, false);
        }
        __syncthreads();
    }
    const int col = nblk + ctile + lrow;
    const float bv = bias ? bias[col] : 0.f;
    #pragma unroll
    for (int r = 0; r < 8; ++r) {
        int row = m0 + rtile + r + (hi ? 8 : 0);
        if (row < M) Y[(long long)row * Ncols + col] = acc[r] + bv;
    }
}

// ---------- self-loop attr precompute: S = segsum(edge_attr,dst)/max(cnt,1) ----------
__global__ void k_scatter_S(const int* dst, const float* eattr, float* S, float* cnt, long long E) {
    long long e = (long long)blockIdx.x * blockDim.x + threadIdx.x;
    if (e >= E) return;
    int d = dst[e];
    atomicAdd(&cnt[d], 1.f);
    #pragma unroll
    for (int f = 0; f < FEDGE; ++f) atomicAdd(&S[(long long)d * FEDGE + f], eattr[e * FEDGE + f]);
}
__global__ void k_norm_S(float* S, const float* cnt, long long n16) {
    long long i = (long long)blockIdx.x * blockDim.x + threadIdx.x;
    if (i >= n16) return;
    S[i] /= fmaxf(cnt[i >> 4], 1.f);
}

// ---------- per-layer attention folding ----------
// Ve[k,h] = sum_c We[k, h*32+c] * att_edge[h,c]   (256 x 8)
__global__ void k_fold_Ve(const float* We, const float* att_edge, float* Ve) {
    int i = blockIdx.x * blockDim.x + threadIdx.x;
    if (i >= HID * HEADS) return;
    int k = i >> 3, h = i & 7;
    float s = 0.f;
    for (int c = 0; c < CCH; ++c) s += We[(long long)k * HID + h * CCH + c] * att_edge[h * CCH + c];
    Ve[k * HEADS + h] = s;
}
// M[f,h] = sum_k edge_w[f,k]*Ve[k,h];  c[h] = sum_k edge_b[k]*Ve[k,h]
__global__ void k_fold_Mc(const float* edge_w, const float* edge_b, const float* Ve,
                          float* Mf, float* cf) {
    int t = threadIdx.x;
    if (t < FEDGE * HEADS) {
        int f = t >> 3, h = t & 7;
        float s = 0.f;
        for (int k = 0; k < HID; ++k) s += edge_w[f * HID + k] * Ve[k * HEADS + h];
        Mf[f * HEADS + h] = s;
    } else if (t < FEDGE * HEADS + HEADS) {
        int h = t - FEDGE * HEADS;
        float s = 0.f;
        for (int k = 0; k < HID; ++k) s += edge_b[k] * Ve[k * HEADS + h];
        cf[h] = s;
    }
}

// ---------- attention node-side: a_src/a_dst, self-loop raw, init amax/denom ----------
__global__ void k_attn_node(const float* __restrict__ Hp, const float* att_src, const float* att_dst,
                            const float* S, const float* cnt, const float* Mf, const float* cf,
                            float* asrc, float* adst, float* rawself,
                            unsigned* amaxenc, float* denom, long long N) {
    long long i = (long long)blockIdx.x * blockDim.x + threadIdx.x;
    if (i >= N * HEADS) return;
    long long n = i >> 3;
    int h = (int)(i & 7);
    const float* Hr = Hp + n * HID + h * CCH;
    float as = 0.f, ad = 0.f;
    #pragma unroll
    for (int c = 0; c < CCH; ++c) {
        float v = Hr[c];
        as += v * att_src[h * CCH + c];
        ad += v * att_dst[h * CCH + c];
    }
    asrc[i] = as; adst[i] = ad;
    float ae = (cnt[n] > 0.f) ? cf[h] : 0.f;   // loop_attr==0 for isolated dst
    #pragma unroll
    for (int f = 0; f < FEDGE; ++f) ae += S[n * FEDGE + f] * Mf[f * HEADS + h];
    float raw = lrelu(as + ad + ae);
    rawself[i] = raw;
    amaxenc[i] = f32_enc(raw);   // self loop always exists -> valid init of segment max
    denom[i] = 0.f;
}

// ---------- edge passes ----------
__global__ void k_edge_pass1(const int* src, const int* dst, const float* eattr,
                             const float* Mf, const float* cf,
                             const float* asrc, const float* adst,
                             float* exE, unsigned* amaxenc, long long E) {
    long long i = (long long)blockIdx.x * blockDim.x + threadIdx.x;
    if (i >= E * HEADS) return;
    long long e = i >> 3;
    int h = (int)(i & 7);
    int s = src[e], d = dst[e];
    float ae = cf[h];
    #pragma unroll
    for (int f = 0; f < FEDGE; ++f) ae += eattr[e * FEDGE + f] * Mf[f * HEADS + h];
    float raw = lrelu(asrc[(long long)s * HEADS + h] + adst[(long long)d * HEADS + h] + ae);
    exE[i] = raw;
    atomicMax(&amaxenc[(long long)d * HEADS + h], f32_enc(raw));
}
__global__ void k_edge_pass2(const int* dst, float* exE, const unsigned* amaxenc,
                             float* denom, long long E) {
    long long i = (long long)blockIdx.x * blockDim.x + threadIdx.x;
    if (i >= E * HEADS) return;
    long long e = i >> 3;
    int h = (int)(i & 7);
    int d = dst[e];
    float ex = expf(exE[i] - f32_dec(amaxenc[(long long)d * HEADS + h]));
    exE[i] = ex;
    atomicAdd(&denom[(long long)d * HEADS + h], ex);
}
__global__ void k_self_pass2(float* rawself, const unsigned* amaxenc, float* denom, long long N8) {
    long long i = (long long)blockIdx.x * blockDim.x + threadIdx.x;
    if (i >= N8) return;
    float ex = expf(rawself[i] - f32_dec(amaxenc[i]));
    rawself[i] = ex;
    atomicAdd(&denom[i], ex);
}
// self-loop message initializes the aggregation buffer (plain store)
__global__ void k_self_pass3(const float* Hp, const float* exself, const float* denom,
                             float* outagg, long long N) {
    long long i = (long long)blockIdx.x * blockDim.x + threadIdx.x;
    if (i >= N * HID) return;
    long long n = i >> 8;
    int h = (int)((i & 255) >> 5);
    float a = exself[n * HEADS + h] / fmaxf(denom[n * HEADS + h], 1e-16f);
    outagg[i] = Hp[i] * a;
}
__global__ void k_edge_pass3(const int* src, const int* dst, const float* Hp,
                             const float* exE, const float* denom,
                             float* outagg, long long E) {
    long long t = (long long)blockIdx.x * blockDim.x + threadIdx.x;
    if (t >= E * 64) return;
    long long e = t >> 6;
    int j = (int)(t & 63);
    int ch0 = j * 4, h = ch0 >> 5;
    int s = src[e], d = dst[e];
    float al = exE[e * HEADS + h] / fmaxf(denom[(long long)d * HEADS + h], 1e-16f);
    #pragma unroll
    for (int q = 0; q < 4; ++q) {
        int ch = ch0 + q;
        atomicAdd(&outagg[(long long)d * HID + ch], Hp[(long long)s * HID + ch] * al);
    }
}

// ---------- batch norm (two-pass with atomics) ----------
__global__ void k_bn_stats(const float* X, const float* bias, float* sum, float* sq,
                           int rows, int cols) {
    int ch = threadIdx.x;
    int r0 = blockIdx.x * 32;
    int r1 = r0 + 32; if (r1 > rows) r1 = rows;
    float bv = bias ? bias[ch] : 0.f;
    float s = 0.f, q = 0.f;
    for (int r = r0; r < r1; ++r) {
        float v = X[(long long)r * cols + ch] + bv;
        s += v; q += v * v;
    }
    atomicAdd(&sum[ch], s);
    atomicAdd(&sq[ch], q);
}
__global__ void k_bn_apply(const float* X, const float* bias, const float* sum, const float* sq,
                           const float* g, const float* b, const float* res,
                           float* Y, int rows, int cols) {
    long long i = (long long)blockIdx.x * blockDim.x + threadIdx.x;
    if (i >= (long long)rows * cols) return;
    int ch = (int)(i % cols);
    float mu = sum[ch] / rows;
    float var = sq[ch] / rows - mu * mu;
    float v = X[i] + (bias ? bias[ch] : 0.f);
    v = (v - mu) * rsqrtf(var + EPS) * g[ch] + b[ch];
    v = fmaxf(v, 0.f);
    if (res) v += res[i];
    Y[i] = v;
}

// ---------- readout ----------
__global__ void k_cntg(const int* batch, float* cntg, long long N) {
    long long n = (long long)blockIdx.x * blockDim.x + threadIdx.x;
    if (n < N) atomicAdd(&cntg[batch[n]], 1.f);
}
__global__ void k_accum_g(const float* h, const int* batch, float* gsum, unsigned* gmax, long long N) {
    long long i = (long long)blockIdx.x * blockDim.x + threadIdx.x;
    if (i >= N * HID) return;
    long long n = i >> 8;
    int ch = (int)(i & 255);
    int g = batch[n];
    float v = h[i];
    atomicAdd(&gsum[(long long)g * HID + ch], v);
    atomicMax(&gmax[(long long)g * HID + ch], f32_enc(v));
}
__global__ void k_build_z(const float* gsum, const unsigned* gmax, const float* cntg,
                          float* z, int G) {
    long long i = (long long)blockIdx.x * blockDim.x + threadIdx.x;
    if (i >= (long long)G * HID) return;
    int g = (int)(i >> 8), ch = (int)(i & 255);
    float s = gsum[i];
    float c = fmaxf(cntg[g], 1.f);
    z[(long long)g * 768 + ch]       = s / c;
    z[(long long)g * 768 + 256 + ch] = f32_dec(gmax[i]);
    z[(long long)g * 768 + 512 + ch] = s;
}
__global__ void k_head_out(const float* z, const float* wo, const float* bo, float* out, int G) {
    int g = blockIdx.x * blockDim.x + threadIdx.x;
    if (g >= G) return;
    float v = bo[0];
    #pragma unroll
    for (int k = 0; k < 64; ++k) v += z[(long long)g * 64 + k] * wo[k];
    out[g] = v;
}

// ============================= host launcher =============================
static inline unsigned nblk(long long n, int t) { return (unsigned)((n + t - 1) / t); }

extern "C" void kernel_launch(void* const* d_in, const int* in_sizes, int n_in,
                              void* d_out, int out_size, void* d_ws, size_t ws_size,
                              hipStream_t stream) {
    const long long N = in_sizes[0] / 64;
    const long long E = in_sizes[1] / 16;
    const int G = out_size;

    const float* x       = (const float*)d_in[0];
    const float* eattr   = (const float*)d_in[1];
    const float* node_w  = (const float*)d_in[2];
    const float* node_b  = (const float*)d_in[3];
    const float* edge_w  = (const float*)d_in[4];
    const float* edge_b  = (const float*)d_in[5];
    const float* gatW[LAYERS], *gatWe[LAYERS], *gatAS[LAYERS], *gatAD[LAYERS], *gatAE[LAYERS], *gatB[LAYERS];
    for (int l = 0; l < LAYERS; ++l) {
        gatW[l]  = (const float*)d_in[6 + 6 * l + 0];
        gatWe[l] = (const float*)d_in[6 + 6 * l + 1];
        gatAS[l] = (const float*)d_in[6 + 6 * l + 2];
        gatAD[l] = (const float*)d_in[6 + 6 * l + 3];
        gatAE[l] = (const float*)d_in[6 + 6 * l + 4];
        gatB[l]  = (const float*)d_in[6 + 6 * l + 5];
    }
    const float* bnG[LAYERS], *bnB[LAYERS];
    for (int l = 0; l < LAYERS; ++l) {
        bnG[l] = (const float*)d_in[30 + 2 * l + 0];
        bnB[l] = (const float*)d_in[30 + 2 * l + 1];
    }
    const float* hw[3], *hb[3], *hg[3], *hbt[3];
    for (int i = 0; i < 3; ++i) {
        hw[i]  = (const float*)d_in[38 + 4 * i + 0];
        hb[i]  = (const float*)d_in[38 + 4 * i + 1];
        hg[i]  = (const float*)d_in[38 + 4 * i + 2];
        hbt[i] = (const float*)d_in[38 + 4 * i + 3];
    }
    const float* how = (const float*)d_in[50];
    const float* hob = (const float*)d_in[51];
    const int* ei    = (const int*)d_in[52];
    const int* src   = ei;
    const int* dst   = ei + E;
    const int* batch = (const int*)d_in[53];
    float* out = (float*)d_out;

    // ---- workspace carve-out ----
    char* base = (char*)d_ws;
    size_t off = 0;
    auto alloc = [&](size_t bytes) -> void* {
        size_t a = (off + 255) & ~(size_t)255;
        void* p = base + a;
        off = a + bytes;
        return p;
    };
    float*    hA      = (float*)alloc((size_t)N * HID * 4);
    float*    hB      = (float*)alloc((size_t)N * HID * 4);
    float*    Hp      = (float*)alloc((size_t)N * HID * 4);
    float*    outagg  = (float*)alloc((size_t)N * HID * 4);
    _Float16* A16     = (_Float16*)alloc((size_t)N * HID * 2);
    _Float16* B16     = (_Float16*)alloc((size_t)768 * 256 * 2);
    float*    exE     = (float*)alloc((size_t)E * HEADS * 4);
    float*    rawself = (float*)alloc((size_t)N * HEADS * 4);
    unsigned* amaxenc = (unsigned*)alloc((size_t)N * HEADS * 4);
    float*    denom   = (float*)alloc((size_t)N * HEADS * 4);
    float*    asrc    = (float*)alloc((size_t)N * HEADS * 4);
    float*    adst    = (float*)alloc((size_t)N * HEADS * 4);
    float*    Sbuf    = (float*)alloc((size_t)N * FEDGE * 4);
    float*    cnt     = (float*)alloc((size_t)N * 4);
    float*    Ve      = (float*)alloc((size_t)HID * HEADS * 4);
    float*    Mf      = (float*)alloc((size_t)FEDGE * HEADS * 4);
    float*    cf      = (float*)alloc((size_t)HEADS * 4);
    float*    bnsum   = (float*)alloc(768 * 4);
    float*    bnsq    = (float*)alloc(768 * 4);
    float*    gsum    = (float*)alloc((size_t)G * HID * 4);
    unsigned* gmax    = (unsigned*)alloc((size_t)G * HID * 4);
    float*    cntg    = (float*)alloc((size_t)G * 4);
    float*    zA      = (float*)alloc((size_t)G * 768 * 4);
    float*    zB      = (float*)alloc((size_t)G * 768 * 4);
    (void)ws_size; (void)n_in;

    // encoded -FLT_MAX for max-init
    float negmax = -3.402823466e38f;
    unsigned nm_bits; memcpy(&nm_bits, &negmax, 4);
    unsigned nm_enc = (nm_bits & 0x80000000u) ? ~nm_bits : (nm_bits | 0x80000000u);

    auto gemm = [&](const _Float16* A, const _Float16* B, const float* bias, float* Y,
                    int M, int K, int Nc) {
        dim3 grid((M + 31) / 32, Nc / 64);
        k_gemm_wmma_f16<<<grid, 256, 0, stream>>>(A, B, bias, Y, M, K, Nc);
    };

    // ---- S = segsum(edge_attr,dst)/max(cnt,1), cnt ----
    k_zero_f32<<<nblk(N * FEDGE, 256), 256, 0, stream>>>(Sbuf, N * FEDGE);
    k_zero_f32<<<nblk(N, 256), 256, 0, stream>>>(cnt, N);
    k_scatter_S<<<nblk(E, 256), 256, 0, stream>>>(dst, eattr, Sbuf, cnt, E);
    k_norm_S<<<nblk(N * FEDGE, 256), 256, 0, stream>>>(Sbuf, cnt, N * FEDGE);

    // ---- node embedding: hA = x @ node_w + node_b  (N x 64 @ 64 x 256) ----
    k_f32_to_f16<<<nblk(N * 64, 256), 256, 0, stream>>>(x, A16, N * 64);
    k_f32_to_f16<<<nblk(64 * 256, 256), 256, 0, stream>>>(node_w, B16, 64 * 256);
    gemm(A16, B16, node_b, hA, (int)N, 64, HID);

    // ---- GAT layers ----
    float* cur = hA;
    float* nxt = hB;
    for (int l = 0; l < LAYERS; ++l) {
        k_fold_Ve<<<nblk(HID * HEADS, 256), 256, 0, stream>>>(gatWe[l], gatAE[l], Ve);
        k_fold_Mc<<<1, 256, 0, stream>>>(edge_w, edge_b, Ve, Mf, cf);

        k_f32_to_f16<<<nblk(N * HID, 256), 256, 0, stream>>>(cur, A16, N * HID);
        k_f32_to_f16<<<nblk(HID * HID, 256), 256, 0, stream>>>(gatW[l], B16, HID * HID);
        gemm(A16, B16, nullptr, Hp, (int)N, HID, HID);   // v_wmma path

        k_attn_node<<<nblk(N * HEADS, 256), 256, 0, stream>>>(Hp, gatAS[l], gatAD[l],
            Sbuf, cnt, Mf, cf, asrc, adst, rawself, amaxenc, denom, N);
        k_edge_pass1<<<nblk(E * HEADS, 256), 256, 0, stream>>>(src, dst, eattr, Mf, cf,
            asrc, adst, exE, amaxenc, E);
        k_edge_pass2<<<nblk(E * HEADS, 256), 256, 0, stream>>>(dst, exE, amaxenc, denom, E);
        k_self_pass2<<<nblk(N * HEADS, 256), 256, 0, stream>>>(rawself, amaxenc, denom, N * HEADS);
        k_self_pass3<<<nblk(N * HID, 256), 256, 0, stream>>>(Hp, rawself, denom, outagg, N);
        k_edge_pass3<<<nblk(E * 64, 256), 256, 0, stream>>>(src, dst, Hp, exE, denom, outagg, E);

        k_zero_f32<<<nblk(768, 256), 256, 0, stream>>>(bnsum, 768);
        k_zero_f32<<<nblk(768, 256), 256, 0, stream>>>(bnsq, 768);
        k_bn_stats<<<nblk(N, 32), HID, 0, stream>>>(outagg, gatB[l], bnsum, bnsq, (int)N, HID);
        k_bn_apply<<<nblk(N * HID, 256), 256, 0, stream>>>(outagg, gatB[l], bnsum, bnsq,
            bnG[l], bnB[l], cur, nxt, (int)N, HID);
        float* t = cur; cur = nxt; nxt = t;
    }

    // ---- readout ----
    k_zero_f32<<<nblk((long long)G * HID, 256), 256, 0, stream>>>(gsum, (long long)G * HID);
    k_zero_f32<<<nblk(G, 256), 256, 0, stream>>>(cntg, G);
    k_fill_u32<<<nblk((long long)G * HID, 256), 256, 0, stream>>>(gmax, nm_enc, (long long)G * HID);
    k_cntg<<<nblk(N, 256), 256, 0, stream>>>(batch, cntg, N);
    k_accum_g<<<nblk(N * HID, 256), 256, 0, stream>>>(cur, batch, gsum, gmax, N);
    k_build_z<<<nblk((long long)G * HID, 256), 256, 0, stream>>>(gsum, gmax, cntg, zA, G);

    // ---- head MLP: 768 -> 256 -> 128 -> 64, each: WMMA gemm(+bias) -> BN -> ReLU ----
    int K = 768;
    const int fos[3] = {256, 128, 64};
    for (int i = 0; i < 3; ++i) {
        int fo = fos[i];
        k_f32_to_f16<<<nblk((long long)G * K, 256), 256, 0, stream>>>(zA, A16, (long long)G * K);
        k_f32_to_f16<<<nblk((long long)K * fo, 256), 256, 0, stream>>>(hw[i], B16, (long long)K * fo);
        gemm(A16, B16, hb[i], zB, G, K, fo);
        k_zero_f32<<<nblk(768, 256), 256, 0, stream>>>(bnsum, 768);
        k_zero_f32<<<nblk(768, 256), 256, 0, stream>>>(bnsq, 768);
        k_bn_stats<<<nblk(G, 32), fo, 0, stream>>>(zB, nullptr, bnsum, bnsq, G, fo);
        k_bn_apply<<<nblk((long long)G * fo, 256), 256, 0, stream>>>(zB, nullptr, bnsum, bnsq,
            hg[i], hbt[i], nullptr, zA, G, fo);
        K = fo;
    }
    k_head_out<<<nblk(G, 256), 256, 0, stream>>>(zA, how, hob, out, G);
}
